// QCPINN_33036888441103
// MI455X (gfx1250) — matrix operand order repaired and checked by reading, not compiled, not run on hardware
//
#include <hip/hip_runtime.h>
#include <hip/hip_bf16.h>
#include <math.h>

typedef __attribute__((ext_vector_type(16))) _Float16 v16h;
typedef __attribute__((ext_vector_type(8)))  _Float16 v8h;
typedef __attribute__((ext_vector_type(2)))  _Float16 h2;
typedef __attribute__((ext_vector_type(8)))  float    v8f;

#define ST 40   // LDS row stride in halves (80 B -> 16B-aligned rows, conflict-free)

// ---- workspace byte offsets (all < 16 KB) ----
enum : int {
  WS_B2PRE  = 0,      // pre_w2 in WMMA B-layout, 2 halves: 2*32*16 f16 = 2048 B
  WS_B2POST = 2048,   // post_w2 likewise
  WS_W1L    = 4096,   // pre_w1  per-lane A-layout: 32*16 f32
  WS_B1L    = 6144,   // pre_b1  per-lane A-layout
  WS_PW1A   = 8192,   // post_w1 row 0, per-lane A-layout
  WS_PW1B   = 10240,  // post_w1 row 1
  WS_PB1    = 12288,  // post_b1
  WS_W3PRE  = 14336,  // pre_w3, column-major [2][32] f32, LDS-position-permuted
  WS_W3POST = 14592,  // post_w3 [32] f32, LDS-position-permuted
  WS_GATES  = 14720,  // fused quantum gates: 16 float2
};

struct cpx { float x, y; };
__device__ __forceinline__ cpx Cf(float2 f) { return {f.x, f.y}; }
__device__ __forceinline__ cpx cmul(cpx a, cpx b) {
  return { fmaf(a.x, b.x, -a.y * b.y), fmaf(a.x, b.y, a.y * b.x) };
}
__device__ __forceinline__ cpx cmadd(cpx a, cpx b, cpx c) {
  return { fmaf(a.x, b.x, fmaf(-a.y, b.y, c.x)),
           fmaf(a.x, b.y, fmaf( a.y, b.x, c.y)) };
}

// CDNA5 native V_TANH_F32 (TRANS op); fallback for older toolchains.
__device__ __forceinline__ float tanh_fast(float v) {
#if __has_builtin(__builtin_amdgcn_tanhf)
  return __builtin_amdgcn_tanhf(v);
#else
  float e = __builtin_amdgcn_exp2f(v * 2.8853900817779268f);  // 2*log2(e)
  return fmaf(-2.0f, __builtin_amdgcn_rcpf(e + 1.0f), 1.0f);
#endif
}
// exchange value with lane^16 (SWAPX16: xor=0x10,or=0,and=0x1f -> 0x401F)
__device__ __forceinline__ float swapx16(float v) {
  return __int_as_float(__builtin_amdgcn_ds_swizzle(__float_as_int(v), 0x401F));
}

// LDS column interleave: position p (0..31) <-> original column (p>>1) + 16*(p&1)

// ---------------------------------------------------------------------------
// Prep kernel: pack weights into WMMA lane layouts + build fused quantum gates
// ---------------------------------------------------------------------------
__global__ __launch_bounds__(128) void qcpinn_prep(
    const float* __restrict__ pre_w1, const float* __restrict__ pre_b1,
    const float* __restrict__ pre_w2, const float* __restrict__ pre_w3,
    const float* __restrict__ qw,
    const float* __restrict__ post_w1, const float* __restrict__ post_b1,
    const float* __restrict__ post_w2, const float* __restrict__ post_w3,
    char* __restrict__ ws)
{
  const int tid = threadIdx.x;
  _Float16* B2pre  = (_Float16*)(ws + WS_B2PRE);
  _Float16* B2post = (_Float16*)(ws + WS_B2POST);
  float* w1l  = (float*)(ws + WS_W1L);
  float* b1l  = (float*)(ws + WS_B1L);
  float* pw1a = (float*)(ws + WS_PW1A);
  float* pw1b = (float*)(ws + WS_PW1B);
  float* pb1  = (float*)(ws + WS_PB1);
  float* w3pre  = (float*)(ws + WS_W3PRE);
  float* w3post = (float*)(ws + WS_W3POST);
  float2* gates = (float2*)(ws + WS_GATES);

  // B-layout (16-bit B 32x16): lane ln holds column (ln&15) of half h;
  // element e <-> K = e + 8*((e>>3) + (ln>>4))
  for (int i = tid; i < 2 * 2 * 32 * 16; i += 128) {
    int e   = i & 15;
    int ln  = (i >> 4) & 31;
    int h   = (i >> 9) & 1;
    int m   = i >> 10;
    int k   = e + 8 * ((e >> 3) + (ln >> 4));
    int col = 16 * h + (ln & 15);
    const float* W = m ? post_w2 : pre_w2;
    _Float16*  dst = m ? B2post : B2pre;
    dst[(h * 32 + ln) * 16 + e] = (_Float16)W[k * 32 + col];
  }
  // A-layout K mapping: k = e + 8*((e>>3) + half)
  for (int i = tid; i < 32 * 16; i += 128) {
    int e = i & 15, ln = i >> 4;
    int k = e + 8 * ((e >> 3) + (ln >> 4));
    w1l[i]  = pre_w1[k];
    b1l[i]  = pre_b1[k];
    pw1a[i] = post_w1[k];
    pw1b[i] = post_w1[32 + k];
    pb1[i]  = post_b1[k];
  }
  // layer-3 / final dot weights, permuted to the interleaved LDS position order
  for (int i = tid; i < 64; i += 128) {
    int c = i >> 5, p = i & 31;
    int k = (p >> 1) + 16 * (p & 1);
    w3pre[c * 32 + p] = pre_w3[k * 2 + c];   // pre_w3 is [32,2] row-major
  }
  for (int i = tid; i < 32; i += 128) {
    int k = (i >> 1) + 16 * (i & 1);
    w3post[i] = post_w3[k];
  }

  // fused single-qubit gates g = RZ(az) @ RY(ay) @ RX(ax), complex 2x2
  if (tid < 4) {
    int l = tid >> 1, qb = tid & 1;
    const float* a = qw + (l * 2 + qb) * 3;
    float cx = cosf(0.5f * a[0]), sx = sinf(0.5f * a[0]);
    float cy = cosf(0.5f * a[1]), sy = sinf(0.5f * a[1]);
    float cz = cosf(0.5f * a[2]), sz = sinf(0.5f * a[2]);
    // m = RY @ RX
    float m00x =  cy * cx, m00y =  sy * sx;
    float m01x = -sy * cx, m01y = -cy * sx;
    float m10x =  sy * cx, m10y = -cy * sx;
    float m11x =  cy * cx, m11y = -sy * sx;
    // row0 *= (cz,-sz); row1 *= (cz,+sz)
    float2* d = gates + l * 8 + qb * 4;
    d[0] = make_float2(cz * m00x + sz * m00y, cz * m00y - sz * m00x);
    d[1] = make_float2(cz * m01x + sz * m01y, cz * m01y - sz * m01x);
    d[2] = make_float2(cz * m10x - sz * m10y, cz * m10y + sz * m10x);
    d[3] = make_float2(cz * m11x - sz * m11y, cz * m11y + sz * m11x);
  }
}

// ---------------------------------------------------------------------------
// Main kernel: one 16-row tile per wave; 4x v_wmma_f32_16x16x32_f16 per tile
// ---------------------------------------------------------------------------
__global__ __launch_bounds__(256) void qcpinn_main(
    const float* __restrict__ x,
    const float* __restrict__ pre_b2, const float* __restrict__ pre_b3,
    const float* __restrict__ post_b2, const float* __restrict__ post_b3,
    const char* __restrict__ ws,
    float* __restrict__ out, int N, int nTiles)
{
  __shared__ _Float16 lds[8 * 16 * ST];
  const int lane = threadIdx.x & 31;
  const int wv   = threadIdx.x >> 5;
  const int tile = blockIdx.x * 8 + wv;
  if (tile >= nTiles) return;               // uniform per wave: EXEC stays all-1s
  const int r    = lane & 15;
  const int half = lane >> 4;
  const int col  = lane & 15;
  _Float16* L = lds + wv * (16 * ST);       // per-wave private LDS tile

  const _Float16* B2pre  = (const _Float16*)(ws + WS_B2PRE);
  const _Float16* B2post = (const _Float16*)(ws + WS_B2POST);
  const float* w1l  = (const float*)(ws + WS_W1L)  + lane * 16;
  const float* b1l  = (const float*)(ws + WS_B1L)  + lane * 16;
  const float* pw1a = (const float*)(ws + WS_PW1A) + lane * 16;
  const float* pw1b = (const float*)(ws + WS_PW1B) + lane * 16;
  const float* pb1  = (const float*)(ws + WS_PB1)  + lane * 16;
  const float* w3q  = (const float*)(ws + WS_W3PRE)  + half * 32;
  const float* w3u  = (const float*)(ws + WS_W3POST) + half * 16;
  const float2* Gp  = (const float2*)(ws + WS_GATES);

  const int rowBase = tile * 16;
  int rowc = rowBase + r; if (rowc >= N) rowc = N - 1;
  const float xv = x[rowc];

  // ---- pre-MLP layer 1 (elementwise; built directly in A-layout) ----
  v16h A;
  #pragma unroll
  for (int e = 0; e < 16; ++e)
    A[e] = (_Float16)tanh_fast(fmaf(xv, w1l[e], b1l[e]));

  v16h B0 = *(const v16h*)(B2pre + lane * 16);
  v16h B1 = *(const v16h*)(B2pre + (32 + lane) * 16);
  v8f zero8 = {};
  v8f c0 = __builtin_amdgcn_wmma_f32_16x16x32_f16(false, A, false, B0, (short)0, zero8, false, false);
  v8f c1 = __builtin_amdgcn_wmma_f32_16x16x32_f16(false, A, false, B1, (short)0, zero8, false, false);

  // bias + tanh; C-layout -> interleaved-column f16 tile in LDS (one b32/lane/row)
  const float bz0 = pre_b2[col], bz1 = pre_b2[col + 16];
  #pragma unroll
  for (int j = 0; j < 8; ++j) {
    int rr = j + 8 * half;
    h2 p; p[0] = (_Float16)tanh_fast(c0[j] + bz0);
          p[1] = (_Float16)tanh_fast(c1[j] + bz1);
    *(h2*)(L + rr * ST + 2 * col) = p;      // positions 2c, 2c+1 <-> cols c, c+16
  }
  __builtin_amdgcn_wave_barrier();
  asm volatile("s_wait_dscnt 0" ::: "memory");   // DS in-order within wave

  // ---- pre-MLP layer 3: q[r][half] = h2[r,:] . w3[:,half] + b3[half] ----
  const v8h* rowp = (const v8h*)(L + r * ST);
  v8h R0 = rowp[0], R1 = rowp[1], R2 = rowp[2], R3 = rowp[3];
  float q = pre_b3[half];
  #pragma unroll
  for (int k = 0; k < 8; ++k) q = fmaf((float)R0[k], w3q[k],      q);
  #pragma unroll
  for (int k = 0; k < 8; ++k) q = fmaf((float)R1[k], w3q[8 + k],  q);
  #pragma unroll
  for (int k = 0; k < 8; ++k) q = fmaf((float)R2[k], w3q[16 + k], q);
  #pragma unroll
  for (int k = 0; k < 8; ++k) q = fmaf((float)R3[k], w3q[24 + k], q);
  float qo = swapx16(q);
  float z0 = half ? qo : q;     // enc[:,0]
  float z1 = half ? q  : qo;    // enc[:,1]

  // ---- QNN: 2-qubit statevector with precomputed fused gates ----
  float sa, ca, sb, cb;
  __sincosf(0.5f * z0, &sa, &ca);
  __sincosf(0.5f * z1, &sb, &cb);
  cpx s00{ca * cb, 0.f}, s01{ca * sb, 0.f}, s10{sa * cb, 0.f}, s11{sa * sb, 0.f};
  #pragma unroll
  for (int l = 0; l < 2; ++l) {
    cpx g00 = Cf(Gp[l*8+0]), g01 = Cf(Gp[l*8+1]), g10 = Cf(Gp[l*8+2]), g11 = Cf(Gp[l*8+3]);
    cpx t00 = cmadd(g00, s00, cmul(g01, s10));
    cpx t01 = cmadd(g00, s01, cmul(g01, s11));
    cpx t10 = cmadd(g10, s00, cmul(g11, s10));
    cpx t11 = cmadd(g10, s01, cmul(g11, s11));
    cpx h00 = Cf(Gp[l*8+4]), h01 = Cf(Gp[l*8+5]), h10 = Cf(Gp[l*8+6]), h11 = Cf(Gp[l*8+7]);
    cpx u00 = cmadd(t00, h00, cmul(t01, h01));
    cpx u01 = cmadd(t00, h10, cmul(t01, h11));
    cpx u10 = cmadd(t10, h00, cmul(t11, h01));
    cpx u11 = cmadd(t10, h10, cmul(t11, h11));
    s00 = u00; s01 = u01; s10 = u11; s11 = u10;   // CNOT(ctrl=0,tgt=1)
  }
  float p00 = fmaf(s00.x, s00.x, s00.y * s00.y);
  float p01 = fmaf(s01.x, s01.x, s01.y * s01.y);
  float p10 = fmaf(s10.x, s10.x, s10.y * s10.y);
  float p11 = fmaf(s11.x, s11.x, s11.y * s11.y);
  float zz0 = (p00 + p01) - (p10 + p11);
  float zz1 = (p00 + p10) - (p01 + p11);

  // ---- post-MLP layer 1 (elementwise in A-layout) ----
  v16h A2;
  #pragma unroll
  for (int e = 0; e < 16; ++e)
    A2[e] = (_Float16)tanh_fast(fmaf(zz0, pw1a[e], fmaf(zz1, pw1b[e], pb1[e])));
  v16h D0 = *(const v16h*)(B2post + lane * 16);
  v16h D1 = *(const v16h*)(B2post + (32 + lane) * 16);
  v8f d0 = __builtin_amdgcn_wmma_f32_16x16x32_f16(false, A2, false, D0, (short)0, zero8, false, false);
  v8f d1 = __builtin_amdgcn_wmma_f32_16x16x32_f16(false, A2, false, D1, (short)0, zero8, false, false);

  const float pz0 = post_b2[col], pz1 = post_b2[col + 16];
  #pragma unroll
  for (int j = 0; j < 8; ++j) {
    int rr = j + 8 * half;
    h2 p; p[0] = (_Float16)tanh_fast(d0[j] + pz0);
          p[1] = (_Float16)tanh_fast(d1[j] + pz1);
    *(h2*)(L + rr * ST + 2 * col) = p;
  }
  __builtin_amdgcn_wave_barrier();
  asm volatile("s_wait_dscnt 0" ::: "memory");

  // ---- final 32 -> 1 dot, split across lane pairs ----
  const v8h* up = (const v8h*)(L + r * ST + 16 * half);
  v8h U0 = up[0], U1 = up[1];
  float acc = 0.f;
  #pragma unroll
  for (int k = 0; k < 8; ++k) acc = fmaf((float)U0[k], w3u[k],     acc);
  #pragma unroll
  for (int k = 0; k < 8; ++k) acc = fmaf((float)U1[k], w3u[8 + k], acc);
  float tot = acc + swapx16(acc);
  if (half == 0 && (rowBase + r) < N)
    out[rowBase + r] = tot + post_b3[0];
}

extern "C" void kernel_launch(void* const* d_in, const int* in_sizes, int n_in,
                              void* d_out, int out_size, void* d_ws, size_t ws_size,
                              hipStream_t stream) {
  (void)n_in; (void)out_size; (void)ws_size;
  const float* x         = (const float*)d_in[0];
  const float* pre_w1    = (const float*)d_in[1];
  const float* pre_b1    = (const float*)d_in[2];
  const float* pre_w2    = (const float*)d_in[3];
  const float* pre_b2    = (const float*)d_in[4];
  const float* pre_w3    = (const float*)d_in[5];
  const float* pre_b3    = (const float*)d_in[6];
  const float* q_weights = (const float*)d_in[7];
  const float* post_w1   = (const float*)d_in[8];
  const float* post_b1   = (const float*)d_in[9];
  const float* post_w2   = (const float*)d_in[10];
  const float* post_b2   = (const float*)d_in[11];
  const float* post_w3   = (const float*)d_in[12];
  const float* post_b3   = (const float*)d_in[13];
  float* out = (float*)d_out;
  char*  ws  = (char*)d_ws;
  const int N = in_sizes[0];

  qcpinn_prep<<<1, 128, 0, stream>>>(pre_w1, pre_b1, pre_w2, pre_w3, q_weights,
                                     post_w1, post_b1, post_w2, post_w3, ws);

  const int nTiles = (N + 15) / 16;          // N = 1e6 -> 62500 full tiles
  const int blocks = (nTiles + 7) / 8;       // 8 waves (tiles) per 256-thread block
  qcpinn_main<<<blocks, 256, 0, stream>>>(x, pre_b2, pre_b3, post_b2, post_b3,
                                          ws, out, N, nTiles);
}
